// AttentionWithRelativePosition_8753143349559
// MI455X (gfx1250) — matrix operand order repaired and checked by reading, not compiled, not run on hardware
//
#include <hip/hip_runtime.h>

// MI455X / gfx1250, wave32. All matrix math via v_wmma_f32_16x16x32_bf16
// (fp32 accumulate). fp32 inputs converted to bf16 once up front; GEMM inner
// loop is register-double-buffered so vmem latency hides under WMMA; softmax
// row-sums computed by an extra WMMA against a ones matrix (no shuffle tree).

typedef __attribute__((ext_vector_type(16))) __bf16 v16bf;
typedef __attribute__((ext_vector_type(8)))  __bf16 v8bf;
typedef __attribute__((ext_vector_type(8)))  float  v8f;
typedef __attribute__((ext_vector_type(4)))  float  f4;

#define NH    12
#define NP    197
#define CDIM  768
#define HD    64
#define QKVLD (3 * CDIM)     // 2304
#define BATCH 64
#define MROWS (BATCH * NP)   // 12608 = 394*32, tile-exact

__device__ __forceinline__ __bf16 f2bf(float f) {
  unsigned u = __builtin_bit_cast(unsigned, f);
  u = (u + 0x7fffu + ((u >> 16) & 1u)) >> 16;   // round-to-nearest-even
  unsigned short h = (unsigned short)u;
  return __builtin_bit_cast(__bf16, h);
}

__device__ __forceinline__ v8f wmma_bf16(v16bf a, v16bf b, v8f c) {
  return __builtin_amdgcn_wmma_f32_16x16x32_bf16(false, a, false, b,
                                                 (short)0, c, false, false);
}

// A-operand fragment (16x32 bf16): lane = row (lane&15); 16 K values are
// {c..c+7, c+16..c+23}, c = 8*(lane>>4).  p = row_base + kt + c. Two 16B loads.
__device__ __forceinline__ v16bf ldA(const __bf16* __restrict__ p) {
  v8bf lo = *(const v8bf*)(p);
  v8bf hi = *(const v8bf*)(p + 16);
  return __builtin_shufflevector(lo, hi, 0,1,2,3,4,5,6,7,8,9,10,11,12,13,14,15);
}

// B-operand fragment (32x16 bf16): lane = column (lane&15); 16 K values are
// {c..c+7, c+16..c+23}, c = 8*(lane>>4).  p = col_base + kt + c. Two 16B loads.
__device__ __forceinline__ v16bf ldB(const __bf16* __restrict__ p) {
  v8bf lo = *(const v8bf*)(p);
  v8bf hi = *(const v8bf*)(p + 16);
  return __builtin_shufflevector(lo, hi, 0,1,2,3,4,5,6,7,8,9,10,11,12,13,14,15);
}

// ---------------------------------------------------------------------------
// fp32 -> bf16 bulk convert (n multiple of 8)
// ---------------------------------------------------------------------------
__global__ __launch_bounds__(256) void f32_to_bf16(
    const float* __restrict__ in, __bf16* __restrict__ out, int n) {
  int i = (blockIdx.x * 256 + threadIdx.x) * 8;
  if (i >= n) return;
  f4 a = *(const f4*)(in + i);
  f4 b = *(const f4*)(in + i + 4);
  v8bf v;
  v[0] = f2bf(a[0]); v[1] = f2bf(a[1]); v[2] = f2bf(a[2]); v[3] = f2bf(a[3]);
  v[4] = f2bf(b[0]); v[5] = f2bf(b[1]); v[6] = f2bf(b[2]); v[7] = f2bf(b[3]);
  *(v8bf*)(out + i) = v;
}

// ---------------------------------------------------------------------------
// Dense bias gather: biasH[h][i][j] = table[rel_index[i][j]][h]
// ---------------------------------------------------------------------------
__global__ __launch_bounds__(256) void bias_expand(
    const int* __restrict__ rel, const float* __restrict__ table,
    float* __restrict__ biasH) {
  int idx = blockIdx.x * 256 + threadIdx.x;
  if (idx >= NP * NP) return;
  int t = rel[idx];
#pragma unroll
  for (int h = 0; h < NH; ++h)
    biasH[(size_t)h * NP * NP + idx] = table[t * NH + h];
}

// ---------------------------------------------------------------------------
// Y[M,N] = A[M,K] @ W[N,K]^T (+ bias[N]).  One wave: 32x64 tile =
// 2 A-frags x 4 B-frags -> 8 WMMAs per 32-wide K step on 12 b128 loads.
// Register double-buffered: next step's fragments load while current step's
// WMMAs execute. OUT_BF16 selects bf16 or fp32 output.
// ---------------------------------------------------------------------------
template <bool OUT_BF16>
__global__ __launch_bounds__(32) void gemm_bf16(
    const __bf16* __restrict__ A, const __bf16* __restrict__ W,
    const float* __restrict__ bias, void* __restrict__ Yv,
    int N, int K) {
  const int lane = threadIdx.x;
  const int half = lane >> 4, lq = lane & 15;
  const int cA = half << 3;    // 0 or 8
  const int cB = half << 3;    // 0 or 8
  const int mt = blockIdx.y << 5;   // 32 rows
  const int nt = blockIdx.x << 6;   // 64 cols

  const __bf16* ar0 = A + (size_t)(mt + lq) * K + cA;
  const __bf16* ar1 = A + (size_t)(mt + 16 + lq) * K + cA;
  const __bf16* w0 = W + (size_t)(nt + lq) * K + cB;
  const __bf16* w1 = W + (size_t)(nt + 16 + lq) * K + cB;
  const __bf16* w2 = W + (size_t)(nt + 32 + lq) * K + cB;
  const __bf16* w3 = W + (size_t)(nt + 48 + lq) * K + cB;

  v8f acc00 = {}, acc01 = {}, acc02 = {}, acc03 = {};
  v8f acc10 = {}, acc11 = {}, acc12 = {}, acc13 = {};

  // prologue: fragments for kt = 0
  v16bf a0 = ldA(ar0), a1 = ldA(ar1);
  v16bf b0 = ldB(w0), b1 = ldB(w1), b2 = ldB(w2), b3 = ldB(w3);

  for (int kt = 32; kt < K; kt += 32) {
    // prefetch next step (issues before WMMAs consume current registers)
    v16bf na0 = ldA(ar0 + kt), na1 = ldA(ar1 + kt);
    v16bf nb0 = ldB(w0 + kt), nb1 = ldB(w1 + kt);
    v16bf nb2 = ldB(w2 + kt), nb3 = ldB(w3 + kt);

    acc00 = wmma_bf16(a0, b0, acc00);
    acc01 = wmma_bf16(a0, b1, acc01);
    acc02 = wmma_bf16(a0, b2, acc02);
    acc03 = wmma_bf16(a0, b3, acc03);
    acc10 = wmma_bf16(a1, b0, acc10);
    acc11 = wmma_bf16(a1, b1, acc11);
    acc12 = wmma_bf16(a1, b2, acc12);
    acc13 = wmma_bf16(a1, b3, acc13);

    a0 = na0; a1 = na1;
    b0 = nb0; b1 = nb1; b2 = nb2; b3 = nb3;
  }
  acc00 = wmma_bf16(a0, b0, acc00);
  acc01 = wmma_bf16(a0, b1, acc01);
  acc02 = wmma_bf16(a0, b2, acc02);
  acc03 = wmma_bf16(a0, b3, acc03);
  acc10 = wmma_bf16(a1, b0, acc10);
  acc11 = wmma_bf16(a1, b1, acc11);
  acc12 = wmma_bf16(a1, b2, acc12);
  acc13 = wmma_bf16(a1, b3, acc13);

  float bv0 = bias ? bias[nt + lq]      : 0.f;
  float bv1 = bias ? bias[nt + 16 + lq] : 0.f;
  float bv2 = bias ? bias[nt + 32 + lq] : 0.f;
  float bv3 = bias ? bias[nt + 48 + lq] : 0.f;

  const int m0 = mt + (half << 3);
#pragma unroll
  for (int r = 0; r < 8; ++r) {
    if constexpr (OUT_BF16) {
      __bf16* y0 = (__bf16*)Yv + (size_t)(m0 + r) * N + nt + lq;
      y0[0]  = f2bf(acc00[r] + bv0);
      y0[16] = f2bf(acc01[r] + bv1);
      y0[32] = f2bf(acc02[r] + bv2);
      y0[48] = f2bf(acc03[r] + bv3);
      __bf16* y1 = (__bf16*)Yv + (size_t)(m0 + 16 + r) * N + nt + lq;
      y1[0]  = f2bf(acc10[r] + bv0);
      y1[16] = f2bf(acc11[r] + bv1);
      y1[32] = f2bf(acc12[r] + bv2);
      y1[48] = f2bf(acc13[r] + bv3);
    } else {
      float* y0 = (float*)Yv + (size_t)(m0 + r) * N + nt + lq;
      y0[0]  = acc00[r] + bv0;
      y0[16] = acc01[r] + bv1;
      y0[32] = acc02[r] + bv2;
      y0[48] = acc03[r] + bv3;
      float* y1 = (float*)Yv + (size_t)(m0 + 16 + r) * N + nt + lq;
      y1[0]  = acc10[r] + bv0;
      y1[16] = acc11[r] + bv1;
      y1[32] = acc12[r] + bv2;
      y1[48] = acc13[r] + bv3;
    }
  }
}

// ---------------------------------------------------------------------------
// Flash attention with relative-position bias, all-bf16 operands.
// One wave per (batch, head, 16-query tile); keys in steps of 32
// (197 padded to 224 with -inf mask). Row-sums of exp(P) come from an extra
// WMMA against an all-ones B matrix instead of a shuffle tree.
// ---------------------------------------------------------------------------
__global__ __launch_bounds__(32) void attn_flash(
    const __bf16* __restrict__ qkv, const float* __restrict__ biasH,
    __bf16* __restrict__ aout) {
  const int lane = threadIdx.x;
  const int half = lane >> 4, lq = lane & 15;
  const int cA = half << 3, cB = half << 3;
  const int qt = blockIdx.x << 4;
  const int h  = blockIdx.y;
  const int b  = blockIdx.z;

  __shared__ __bf16 Vt[HD][40];  // V^T [d][key], stride 40 for bank spread
  __shared__ __bf16 Ps[16][40];  // P tile (C-layout scatter -> A-layout load)

  const __bf16* base = qkv + (size_t)b * NP * QKVLD;

  int qm = qt + lq; qm = qm < NP ? qm : NP - 1;   // clamp tail reads
  const __bf16* qrow = base + (size_t)qm * QKVLD + h * HD + cA;
  v16bf qa0 = ldA(qrow);
  v16bf qa1 = ldA(qrow + 32);

  v16bf ones;
#pragma unroll
  for (int e = 0; e < 16; ++e)
    ones[e] = __builtin_bit_cast(__bf16, (unsigned short)0x3F80);  // 1.0bf

  float mi[8], li[8], alpha[8];
  v8f o0 = {}, o1 = {}, o2 = {}, o3 = {};
#pragma unroll
  for (int r = 0; r < 8; ++r) { mi[r] = -3.0e38f; li[r] = 0.f; }
  const float scale = 0.125f;    // 64^-0.5
  const size_t bh = (size_t)h * NP * NP;

  for (int j0 = 0; j0 < 224; j0 += 32) {
    // ---- scores S[16q x 32k] = Q @ K^T (K rows d-contiguous -> ldB) ----
    int ja = j0 + lq, jb = j0 + 16 + lq;
    int jka = ja < NP ? ja : NP - 1;
    int jkb = jb < NP ? jb : NP - 1;
    const __bf16* kpa = base + (size_t)jka * QKVLD + CDIM + h * HD + cB;
    const __bf16* kpb = base + (size_t)jkb * QKVLD + CDIM + h * HD + cB;
    v16bf kf0 = ldB(kpa);
    v16bf kf1 = ldB(kpa + 32);
    v16bf kf2 = ldB(kpb);
    v16bf kf3 = ldB(kpb + 32);
    v8f s0 = {}; s0 = wmma_bf16(qa0, kf0, s0); s0 = wmma_bf16(qa1, kf1, s0);
    v8f s1 = {}; s1 = wmma_bf16(qa0, kf2, s1); s1 = wmma_bf16(qa1, kf3, s1);

    // ---- stage V^T tile to LDS: one key row per lane, pure bf16 moves ----
    int vkey = j0 + lane; vkey = vkey < NP ? vkey : NP - 1;
    const __bf16* vrow = base + (size_t)vkey * QKVLD + 2 * CDIM + h * HD;
    __syncthreads();   // 1-wave WG: orders Vt/Ps reuse across iterations
#pragma unroll
    for (int d = 0; d < HD; d += 8) {
      v8bf vv = *(const v8bf*)(vrow + d);
      Vt[d + 0][lane] = vv[0]; Vt[d + 1][lane] = vv[1];
      Vt[d + 2][lane] = vv[2]; Vt[d + 3][lane] = vv[3];
      Vt[d + 4][lane] = vv[4]; Vt[d + 5][lane] = vv[5];
      Vt[d + 6][lane] = vv[6]; Vt[d + 7][lane] = vv[7];
    }

    // ---- bias + mask + online row-max (wave32 shfl_xor), exp to LDS ----
    bool okA = ja < NP, okB = jb < NP;
#pragma unroll
    for (int r = 0; r < 8; ++r) {
      int i = qt + (half << 3) + r;
      bool oki = i < NP;
      float t0 = (okA && oki) ? s0[r] * scale + biasH[bh + (size_t)i * NP + ja]
                              : -3.0e38f;
      float t1 = (okB && oki) ? s1[r] * scale + biasH[bh + (size_t)i * NP + jb]
                              : -3.0e38f;
      float mloc = fmaxf(t0, t1);
      mloc = fmaxf(mloc, __shfl_xor(mloc, 1));
      mloc = fmaxf(mloc, __shfl_xor(mloc, 2));
      mloc = fmaxf(mloc, __shfl_xor(mloc, 4));
      mloc = fmaxf(mloc, __shfl_xor(mloc, 8));
      float mnew = fmaxf(mi[r], mloc);
      alpha[r] = __expf(mi[r] - mnew);
      mi[r] = mnew;
      Ps[(half << 3) + r][lq]      = f2bf(__expf(t0 - mnew));  // masked -> 0
      Ps[(half << 3) + r][16 + lq] = f2bf(__expf(t1 - mnew));
    }
#pragma unroll
    for (int r = 0; r < 8; ++r) {
      o0[r] *= alpha[r]; o1[r] *= alpha[r];
      o2[r] *= alpha[r]; o3[r] *= alpha[r];
    }
    __syncthreads();

    // ---- P as A-fragment, V^T as B-fragments; +1 WMMA for row sums ----
    v16bf pA  = ldA(&Ps[lq][0] + cA);
    v16bf vb0 = ldB(&Vt[lq][0] + cB);
    v16bf vb1 = ldB(&Vt[16 + lq][0] + cB);
    v16bf vb2 = ldB(&Vt[32 + lq][0] + cB);
    v16bf vb3 = ldB(&Vt[48 + lq][0] + cB);
    o0 = wmma_bf16(pA, vb0, o0);
    o1 = wmma_bf16(pA, vb1, o1);
    o2 = wmma_bf16(pA, vb2, o2);
    o3 = wmma_bf16(pA, vb3, o3);
    v8f ls = {};
    ls = wmma_bf16(pA, ones, ls);   // ls[r] = sum_k P[m,k], replicated over n
#pragma unroll
    for (int r = 0; r < 8; ++r) li[r] = li[r] * alpha[r] + ls[r];
  }

  // ---- normalize + store bf16 [b, i, h*64 + d] ----
#pragma unroll
  for (int r = 0; r < 8; ++r) {
    int i = qt + (half << 3) + r;
    if (i < NP) {
      float inv = 1.0f / li[r];
      __bf16* op = aout + ((size_t)b * NP + i) * CDIM + h * HD + lq;
      op[0]  = f2bf(o0[r] * inv);
      op[16] = f2bf(o1[r] * inv);
      op[32] = f2bf(o2[r] * inv);
      op[48] = f2bf(o3[r] * inv);
    }
  }
}

// ---------------------------------------------------------------------------
extern "C" void kernel_launch(void* const* d_in, const int* in_sizes, int n_in,
                              void* d_out, int out_size, void* d_ws, size_t ws_size,
                              hipStream_t stream) {
  const float* x      = (const float*)d_in[0];   // [64,197,768]
  const float* table  = (const float*)d_in[1];   // [730,12]
  const float* w_qkv  = (const float*)d_in[2];   // [2304,768]
  const float* w_proj = (const float*)d_in[3];   // [768,768]
  const float* b_proj = (const float*)d_in[4];   // [768]
  const int*   rel    = (const int*)d_in[5];     // [197,197]
  float* out = (float*)d_out;

  const int NX  = MROWS * CDIM;       // 9,682,944
  const int NWQ = QKVLD * CDIM;       // 1,769,472
  const int NWP = CDIM * CDIM;        //   589,824

  char* ws = (char*)d_ws;
  size_t off = 0;
  __bf16* x_bf    = (__bf16*)(ws + off); off += (size_t)NX * 2;
  __bf16* wqkv_bf = (__bf16*)(ws + off); off += (size_t)NWQ * 2;
  __bf16* wproj_bf= (__bf16*)(ws + off); off += (size_t)NWP * 2;
  __bf16* qkv_bf  = (__bf16*)(ws + off); off += (size_t)MROWS * QKVLD * 2;
  __bf16* aout_bf = (__bf16*)(ws + off); off += (size_t)MROWS * CDIM * 2;
  float*  biasH   = (float*)(ws + off);  // 12*197*197*4 ~ 1.9 MB

  // one-time fp32 -> bf16 conversions (keeps all GEMM loops conversion-free)
  f32_to_bf16<<<(NX / 8 + 255) / 256, 256, 0, stream>>>(x, x_bf, NX);
  f32_to_bf16<<<(NWQ / 8 + 255) / 256, 256, 0, stream>>>(w_qkv, wqkv_bf, NWQ);
  f32_to_bf16<<<(NWP / 8 + 255) / 256, 256, 0, stream>>>(w_proj, wproj_bf, NWP);
  bias_expand<<<(NP * NP + 255) / 256, 256, 0, stream>>>(rel, table, biasH);

  // qkv = x @ w_qkv^T : [12608, 2304] bf16
  gemm_bf16<true><<<dim3(QKVLD / 64, MROWS / 32), 32, 0, stream>>>(
      x_bf, wqkv_bf, nullptr, qkv_bf, QKVLD, CDIM);
  // flash attention per (b, h, q-tile) -> bf16
  attn_flash<<<dim3((NP + 15) / 16, NH, BATCH), 32, 0, stream>>>(
      qkv_bf, biasH, aout_bf);
  // out = aout @ w_proj^T + b_proj : [12608, 768] fp32
  gemm_bf16<false><<<dim3(CDIM / 64, MROWS / 32), 32, 0, stream>>>(
      aout_bf, wproj_bf, b_proj, out, CDIM, CDIM);
}
